// MoEClassifier_74148315398466
// MI455X (gfx1250) — matrix operand order repaired and checked by reading, not compiled, not run on hardware
//
#include <hip/hip_runtime.h>
#include <hip/hip_bf16.h>
#include <math.h>

// ---------------- problem constants ----------------
#define TN 8192
#define TD 1024
#define TH 4096
#define TE 8
#define TC 8
#define LN_EPS 1e-5f

// ---------------- WMMA types ----------------
typedef __attribute__((ext_vector_type(16))) __bf16 v16bf;
typedef __attribute__((ext_vector_type(8)))  float  v8f;

union FragAB { v16bf v; unsigned u[8]; };

__device__ __forceinline__ float gelu_f(float x) {
    const float k0 = 0.7978845608028654f; // sqrt(2/pi)
    float x3 = x * x * x;
    return 0.5f * x * (1.0f + tanhf(k0 * (x + 0.044715f * x3)));
}

// ---------------- generic helpers ----------------
__global__ void zero_f32_kernel(float* __restrict__ p, long n) {
    long i = (long)blockIdx.x * blockDim.x + threadIdx.x;
    long stride = (long)gridDim.x * blockDim.x;
    for (; i < n; i += stride) p[i] = 0.0f;
}

__global__ void f32_to_bf16_kernel(const float* __restrict__ in,
                                   __bf16* __restrict__ out, long n) {
    long i = (long)blockIdx.x * blockDim.x + threadIdx.x;
    long stride = (long)gridDim.x * blockDim.x;
    for (; i < n; i += stride) out[i] = (__bf16)in[i];
}

// transpose + convert: in [K][N] fp32 (blockIdx.z selects matrix) -> out [N][K] bf16
__global__ __launch_bounds__(256)
void transp_conv_kernel(const float* __restrict__ in, __bf16* __restrict__ out,
                        int K, int N) {
    __shared__ float t[32][33];
    const float* src = in + (size_t)blockIdx.z * K * N;
    __bf16* dst = out + (size_t)blockIdx.z * K * N;
    int k0 = blockIdx.y * 32, n0 = blockIdx.x * 32;
    int tx = threadIdx.x, ty = threadIdx.y;  // 32 x 8
#pragma unroll
    for (int j = 0; j < 4; ++j)
        t[ty + j * 8][tx] = src[(size_t)(k0 + ty + j * 8) * N + n0 + tx];
    __syncthreads();
#pragma unroll
    for (int j = 0; j < 4; ++j)
        dst[(size_t)(n0 + ty + j * 8) * K + k0 + tx] = (__bf16)t[tx][ty + j * 8];
}

// block-level sum reduction (256 threads)
__device__ __forceinline__ float block_reduce_sum(float v, float* sm) {
    int tid = threadIdx.x;
    sm[tid] = v;
    __syncthreads();
    for (int s = 128; s > 0; s >>= 1) {
        if (tid < s) sm[tid] += sm[tid + s];
        __syncthreads();
    }
    float r = sm[0];
    __syncthreads();
    return r;
}

// ---------------- WMMA GEMM ----------------
// C[M,Nn] = epilogue( A[M,K] @ B[K,Nn] + bias[Nn] )
// A row-major bf16; B PRE-TRANSPOSED: Bt[Nn][K] bf16
// EPI 0: store fp32
// EPI 1: store bf16(gelu(val))
// EPI 2: C += gates[m*TE+expert] * val   (fp32 accumulate into Cv)
#define BM 128
#define BN 256
#define BK 64
#define LDSS 72  // BK + 8 pad; row stride 144B = 9*16B -> every chunk 16B-aligned

template <int EPI>
__global__ __launch_bounds__(256)
void gemm_wmma_kernel(const __bf16* __restrict__ A, const __bf16* __restrict__ Bt,
                      const float* __restrict__ bias, void* __restrict__ Cv,
                      const float* __restrict__ gates, int expert,
                      int M, int Nn, int K) {
    extern __shared__ __align__(16) char smem[];
    __bf16* sA = (__bf16*)smem;                     // [2][BM*LDSS]
    __bf16* sB = sA + 2 * BM * LDSS;                // [2][BN*LDSS]

    const int tid  = threadIdx.x;
    const int lane = tid & 31;
    const int wid  = tid >> 5;   // 0..7
    const int wm   = wid >> 2;   // 0..1 : 64-row slab
    const int wn   = wid & 3;    // 0..3 : 64-col slab
    const int rm   = blockIdx.y * BM;
    const int cn   = blockIdx.x * BN;

    v8f acc[4][4];
#pragma unroll
    for (int i = 0; i < 4; ++i)
#pragma unroll
        for (int j = 0; j < 4; ++j)
#pragma unroll
            for (int r = 0; r < 8; ++r) acc[i][j][r] = 0.0f;

    // staging map: rows x 64 cols, 256 threads, chunks of 8 contiguous bf16
    const int st_r = tid >> 3;        // 0..31 (passes of 32 rows)
    const int st_k = (tid & 7) * 8;   // 0..56

    const int nt = K / BK;

    // ---- prologue: stage tile 0 into buffer 0 ----
    {
#pragma unroll
        for (int c = 0; c < 4; ++c) {   // A: 128 rows
            int r = c * 32 + st_r;
            uint4 va = *(const uint4*)(A + (size_t)(rm + r) * K + st_k);
            *(uint4*)&sA[r * LDSS + st_k] = va;
        }
#pragma unroll
        for (int c = 0; c < 8; ++c) {   // B: 256 rows
            int r = c * 32 + st_r;
            uint4 vb = *(const uint4*)(Bt + (size_t)(cn + r) * K + st_k);
            *(uint4*)&sB[r * LDSS + st_k] = vb;
        }
    }
    __syncthreads();

    for (int kt = 0; kt < nt; ++kt) {
        const int cur = kt & 1;

        // ---- stage next tile into the other buffer FIRST (short register
        //      lifetime: staging regs die before the WMMA block; writes to
        //      buf[cur^1] are safe — its last readers synced 1 barrier ago) ----
        if (kt + 1 < nt) {
            const int k0n = (kt + 1) * BK;
            uint4 ra[4], rb[8];
#pragma unroll
            for (int c = 0; c < 4; ++c)
                ra[c] = *(const uint4*)(A + (size_t)(rm + c * 32 + st_r) * K + k0n + st_k);
#pragma unroll
            for (int c = 0; c < 8; ++c)
                rb[c] = *(const uint4*)(Bt + (size_t)(cn + c * 32 + st_r) * K + k0n + st_k);
            __bf16* dA = sA + (cur ^ 1) * BM * LDSS;
            __bf16* dB = sB + (cur ^ 1) * BN * LDSS;
#pragma unroll
            for (int c = 0; c < 4; ++c)
                *(uint4*)&dA[(c * 32 + st_r) * LDSS + st_k] = ra[c];
#pragma unroll
            for (int c = 0; c < 8; ++c)
                *(uint4*)&dB[(c * 32 + st_r) * LDSS + st_k] = rb[c];
        }

        // ---- compute on buffer `cur`: two K=32 steps, 16 WMMAs each ----
        const __bf16* bA = sA + cur * BM * LDSS;
        const __bf16* bB = sB + cur * BN * LDSS;
#pragma unroll
        for (int ks = 0; ks < BK; ks += 32) {
            FragAB b[4];
            const int kbB = ks + (lane >> 4) * 16;
#pragma unroll
            for (int j = 0; j < 4; ++j) {
                int n = wn * 64 + j * 16 + (lane & 15);
                const __bf16* base = &bB[n * LDSS];
#pragma unroll
                for (int t = 0; t < 8; ++t)
                    b[j].u[t] = *(const unsigned*)&base[kbB + 2 * t];
            }
            const int kbA = ks + (lane >> 4) * 16;
#pragma unroll
            for (int i = 0; i < 4; ++i) {
                int m = wm * 64 + i * 16 + (lane & 15);
                const __bf16* base = &bA[m * LDSS];
                FragAB a;
#pragma unroll
                for (int j = 0; j < 4; ++j) {
                    a.u[j]     = *(const unsigned*)&base[kbA + 2 * j];
                    a.u[j + 4] = *(const unsigned*)&base[kbA + 8 + 2 * j];
                }
#pragma unroll
                for (int j = 0; j < 4; ++j)
                    acc[i][j] = __builtin_amdgcn_wmma_f32_16x16x32_bf16(
                        false, a.v, false, b[j].v, (short)0, acc[i][j],
                        false, false);
            }
        }
        __syncthreads();
    }

    // ---- epilogue: C/D layout -> lanes<16: N=lane, M=r; lanes>=16: N=lane-16, M=r+8
#pragma unroll
    for (int i = 0; i < 4; ++i) {
#pragma unroll
        for (int j = 0; j < 4; ++j) {
            int n = cn + wn * 64 + j * 16 + (lane & 15);
            float bn = bias[n];
#pragma unroll
            for (int r = 0; r < 8; ++r) {
                int m = rm + wm * 64 + i * 16 + (lane >> 4) * 8 + r;
                float val = acc[i][j][r] + bn;
                if (EPI == 0) {
                    ((float*)Cv)[(size_t)m * Nn + n] = val;
                } else if (EPI == 1) {
                    ((__bf16*)Cv)[(size_t)m * Nn + n] = (__bf16)gelu_f(val);
                } else {
                    float g = gates[(size_t)m * TE + expert];
                    ((float*)Cv)[(size_t)m * Nn + n] += g * val;
                }
            }
        }
    }
}

// ---------------- LayerNorm (one block per row, 256 thr) ----------------
__global__ __launch_bounds__(256)
void ln_kernel(const float* __restrict__ in, const float* __restrict__ g,
               const float* __restrict__ b, float* __restrict__ out_f32,
               __bf16* __restrict__ out_bf16) {
    __shared__ float red[256];
    int row = blockIdx.x, tid = threadIdx.x;
    const float* x = in + (size_t)row * TD;
    float s = 0.f, s2 = 0.f;
    for (int c = tid; c < TD; c += 256) {
        float v = x[c];
        s += v; s2 += v * v;
    }
    float sum  = block_reduce_sum(s, red);
    float sum2 = block_reduce_sum(s2, red);
    float mu  = sum / TD;
    float var = sum2 / TD - mu * mu;
    float rs  = rsqrtf(var + LN_EPS);
    for (int c = tid; c < TD; c += 256) {
        float y = (x[c] - mu) * rs * g[c] + b[c];
        out_f32[(size_t)row * TD + c]  = y;
        out_bf16[(size_t)row * TD + c] = (__bf16)y;
    }
}

// ---------------- router: softmax -> top2 -> renormalized gates ----------------
__global__ __launch_bounds__(256)
void router_kernel(const float* __restrict__ hidden, const float* __restrict__ Wg,
                   float* __restrict__ gates) {
    __shared__ float red[256];
    __shared__ float logits[TE];
    int row = blockIdx.x, tid = threadIdx.x;
    const float* h = hidden + (size_t)row * TD;
    float p[TE];
#pragma unroll
    for (int e = 0; e < TE; ++e) p[e] = 0.f;
    for (int c = tid; c < TD; c += 256) {
        float hv = h[c];
        const float* w = Wg + (size_t)c * TE;
#pragma unroll
        for (int e = 0; e < TE; ++e) p[e] += hv * w[e];
    }
#pragma unroll
    for (int e = 0; e < TE; ++e) {
        float t = block_reduce_sum(p[e], red);
        if (tid == 0) logits[e] = t;
    }
    __syncthreads();
    if (tid == 0) {
        float mx = logits[0];
        for (int e = 1; e < TE; ++e) mx = fmaxf(mx, logits[e]);
        float pr[TE], Z = 0.f;
        for (int e = 0; e < TE; ++e) { pr[e] = __expf(logits[e] - mx); Z += pr[e]; }
        for (int e = 0; e < TE; ++e) pr[e] /= Z;
        int i0 = 0;
        for (int e = 1; e < TE; ++e) if (pr[e] > pr[i0]) i0 = e;
        int i1 = (i0 == 0) ? 1 : 0;
        for (int e = 0; e < TE; ++e)
            if (e != i0 && pr[e] > pr[i1]) i1 = e;
        float denom = pr[i0] + pr[i1];
        for (int e = 0; e < TE; ++e) {
            float gv = (e == i0) ? pr[i0] : (e == i1 ? pr[i1] : 0.f);
            gates[(size_t)row * TE + e] = gv / denom;
        }
    }
}

// ---------------- final: LN(LN(hidden+mixed)) @ Wc + bc ----------------
__global__ __launch_bounds__(256)
void final_kernel(const float* __restrict__ hidden, const float* __restrict__ mixed,
                  const float* __restrict__ g_moe, const float* __restrict__ b_moe,
                  const float* __restrict__ g_out, const float* __restrict__ b_out,
                  const float* __restrict__ Wc, const float* __restrict__ bc,
                  float* __restrict__ out) {
    __shared__ float red[256];
    __shared__ float buf[TD];
    int row = blockIdx.x, tid = threadIdx.x;
    float s = 0.f, s2 = 0.f;
    for (int c = tid; c < TD; c += 256) {
        float v = hidden[(size_t)row * TD + c] + mixed[(size_t)row * TD + c];
        buf[c] = v;
        s += v; s2 += v * v;
    }
    __syncthreads();
    float sum = block_reduce_sum(s, red), sum2 = block_reduce_sum(s2, red);
    float mu = sum / TD, rs = rsqrtf(sum2 / TD - mu * mu + LN_EPS);
    s = 0.f; s2 = 0.f;
    for (int c = tid; c < TD; c += 256) {
        float y = (buf[c] - mu) * rs * g_moe[c] + b_moe[c];
        buf[c] = y;
        s += y; s2 += y * y;
    }
    __syncthreads();
    sum = block_reduce_sum(s, red); sum2 = block_reduce_sum(s2, red);
    mu = sum / TD; rs = rsqrtf(sum2 / TD - mu * mu + LN_EPS);
    for (int c = tid; c < TD; c += 256)
        buf[c] = (buf[c] - mu) * rs * g_out[c] + b_out[c];
    __syncthreads();
    float p[TC];
#pragma unroll
    for (int k = 0; k < TC; ++k) p[k] = 0.f;
    for (int c = tid; c < TD; c += 256) {
        float z = buf[c];
        const float* w = Wc + (size_t)c * TC;
#pragma unroll
        for (int k = 0; k < TC; ++k) p[k] += z * w[k];
    }
#pragma unroll
    for (int k = 0; k < TC; ++k) {
        float t = block_reduce_sum(p[k], red);
        if (tid == 0) out[(size_t)row * TC + k] = t + bc[k];
    }
}

// ---------------- host launch ----------------
extern "C" void kernel_launch(void* const* d_in, const int* in_sizes, int n_in,
                              void* d_out, int out_size, void* d_ws, size_t ws_size,
                              hipStream_t stream) {
    (void)in_sizes; (void)n_in; (void)out_size; (void)ws_size;
    const float* x     = (const float*)d_in[0];
    const float* Wp    = (const float*)d_in[1];
    const float* bp    = (const float*)d_in[2];
    const float* g_in  = (const float*)d_in[3];
    const float* b_in  = (const float*)d_in[4];
    const float* Wg    = (const float*)d_in[5];
    const float* W1    = (const float*)d_in[6];
    const float* b1    = (const float*)d_in[7];
    const float* W2    = (const float*)d_in[8];
    const float* b2    = (const float*)d_in[9];
    const float* g_moe = (const float*)d_in[10];
    const float* b_moe = (const float*)d_in[11];
    const float* g_out = (const float*)d_in[12];
    const float* b_out = (const float*)d_in[13];
    const float* Wc    = (const float*)d_in[14];
    const float* bc    = (const float*)d_in[15];
    float* out = (float*)d_out;

    // workspace layout (256B aligned)
    char* ws = (char*)d_ws;
    size_t off = 0;
    auto take = [&](size_t bytes) {
        char* p = ws + off;
        off = (off + bytes + 255) & ~(size_t)255;
        return p;
    };
    __bf16* WpT     = (__bf16*)take((size_t)TD * TD * 2);        // Wp^T  [D][D]
    __bf16* W1T     = (__bf16*)take((size_t)TE * TD * TH * 2);   // per-e [H][D]
    __bf16* W2T     = (__bf16*)take((size_t)TE * TH * TD * 2);   // per-e [D][H]
    __bf16* xB      = (__bf16*)take((size_t)TN * TD * 2);        // x as bf16
    float*  hid_pre = (float*)take((size_t)TN * TD * 4);         // reused as `mixed`
    float*  hidden  = (float*)take((size_t)TN * TD * 4);
    __bf16* hiddenB = (__bf16*)take((size_t)TN * TD * 2);
    __bf16* h1B     = (__bf16*)take((size_t)TN * TH * 2);
    float*  gates   = (float*)take((size_t)TN * TE * 4);
    float*  mixed   = hid_pre;

    const size_t gemm_lds = (size_t)2 * (BM + BN) * LDSS * sizeof(__bf16); // 108 KB

    // 1) transpose + convert weights fp32 -> bf16; convert x -> bf16
    {
        dim3 blk(32, 8);
        dim3 gWp(TD / 32, TD / 32, 1);
        transp_conv_kernel<<<gWp, blk, 0, stream>>>(Wp, WpT, TD, TD);
        dim3 gW1(TH / 32, TD / 32, TE);
        transp_conv_kernel<<<gW1, blk, 0, stream>>>(W1, W1T, TD, TH);
        dim3 gW2(TD / 32, TH / 32, TE);
        transp_conv_kernel<<<gW2, blk, 0, stream>>>(W2, W2T, TH, TD);
        f32_to_bf16_kernel<<<2048, 256, 0, stream>>>(x, xB, (long)TN * TD);
    }

    // 2) hidden_pre = x @ Wp + bp
    {
        dim3 grid(TD / BN, TN / BM);
        gemm_wmma_kernel<0><<<grid, 256, gemm_lds, stream>>>(
            xB, WpT, bp, hid_pre, nullptr, 0, TN, TD, TD);
    }
    // 3) hidden = LN(hidden_pre) (fp32 + bf16 copies)
    ln_kernel<<<TN, 256, 0, stream>>>(hid_pre, g_in, b_in, hidden, hiddenB);
    // 4) router gates
    router_kernel<<<TN, 256, 0, stream>>>(hidden, Wg, gates);
    // 5) mixed = 0 (aliases hid_pre, now dead)
    zero_f32_kernel<<<2048, 256, 0, stream>>>(mixed, (long)TN * TD);

    // 6) expert FFNs
    for (int e = 0; e < TE; ++e) {
        dim3 grid1(TH / BN, TN / BM);
        gemm_wmma_kernel<1><<<grid1, 256, gemm_lds, stream>>>(
            hiddenB, W1T + (size_t)e * TD * TH, b1 + (size_t)e * TH,
            h1B, nullptr, 0, TN, TH, TD);
        dim3 grid2(TD / BN, TN / BM);
        gemm_wmma_kernel<2><<<grid2, 256, gemm_lds, stream>>>(
            h1B, W2T + (size_t)e * TH * TD, b2 + (size_t)e * TD,
            mixed, gates, e, TN, TD, TH);
    }

    // 7) double LN + classifier
    final_kernel<<<TN, 256, 0, stream>>>(hidden, mixed, g_moe, b_moe,
                                         g_out, b_out, Wc, bc, out);
}
